// MistralAudioCodebook_88656714924740
// MI455X (gfx1250) — matrix operand order repaired and checked by reading, compiled, not run on hardware
//
#include <hip/hip_runtime.h>
#include <hip/hip_bf16.h>
#include <math.h>

// Problem constants (from reference)
#define BB   8
#define DD   292
#define TT   4096
#define SEM  256
#define NC   8192
#define ACD  (DD - SEM)       // 36
#define EPSV 1e-5f
#define LEVELS 21

typedef __bf16 bf8v  __attribute__((ext_vector_type(8)));
typedef __bf16 bf16v __attribute__((ext_vector_type(16)));
typedef float  v8f   __attribute__((ext_vector_type(8)));

__device__ __forceinline__ bf16v cat8(bf8v a, bf8v b) {
  return __builtin_shufflevector(a, b, 0,1,2,3,4,5,6,7,8,9,10,11,12,13,14,15);
}

// s_wait_asynccnt 0 : all of this wave's async global->LDS transfers complete
__device__ __forceinline__ void wait_async0() {
#if __has_builtin(__builtin_amdgcn_s_wait_asynccnt)
  __builtin_amdgcn_s_wait_asynccnt(0);
#else
  asm volatile("s_wait_asynccnt 0x0" ::: "memory");
#endif
}

// GLOBAL_LOAD_ASYNC_TO_LDS_B128 (GVS mode): LDS[lds_byte] <= 16B @ sbase + gvoff
__device__ __forceinline__ void async_ld16(unsigned lds_byte, unsigned gvoff,
                                           unsigned long long sbase) {
  asm volatile("global_load_async_to_lds_b128 %0, %1, %2"
               :
               : "v"(lds_byte), "v"(gvoff), "s"(sbase)
               : "memory");
}

// ---------------------------------------------------------------------------
// Kernel 1: emb = embedding_sum / max(usage, eps); split into bf16 hi/lo
// planes for split-precision WMMA; e2[n] = sum_k emb^2 (exact f32).
// ---------------------------------------------------------------------------
__global__ __launch_bounds__(SEM)
void prep_emb_kernel(const float* __restrict__ emb_sum,
                     const float* __restrict__ usage,
                     __bf16* __restrict__ ehi, __bf16* __restrict__ elo,
                     float* __restrict__ e2) {
  __shared__ float partial[8];
  const int n = blockIdx.x;
  const int k = threadIdx.x;
  const float u = fmaxf(usage[n], EPSV);
  const float e = emb_sum[(size_t)n * SEM + k] / u;
  const __bf16 h = (__bf16)e;
  const __bf16 l = (__bf16)(e - (float)h);
  ehi[(size_t)n * SEM + k] = h;
  elo[(size_t)n * SEM + k] = l;

  float s = e * e;
  for (int off = 16; off > 0; off >>= 1) s += __shfl_down(s, off, 32);
  if ((threadIdx.x & 31) == 0) partial[threadIdx.x >> 5] = s;
  __syncthreads();
  if (threadIdx.x == 0) {
    float t = 0.f;
    for (int i = 0; i < 8; ++i) t += partial[i];
    e2[n] = t;
  }
}

// ---------------------------------------------------------------------------
// Kernel 2: FSQ channels (elementwise)
// ---------------------------------------------------------------------------
__global__ __launch_bounds__(256)
void ac_kernel(const float* __restrict__ x,
               float* __restrict__ out_codes, float* __restrict__ out_recon) {
  const int total = BB * ACD * TT;
  int idx = blockIdx.x * blockDim.x + threadIdx.x;
  if (idx >= total) return;
  const int t = idx % TT;
  const int d = (idx / TT) % ACD;
  const int b = idx / (TT * ACD);
  const float v = x[(size_t)b * DD * TT + (size_t)(SEM + d) * TT + t];
  const float a = tanhf(v);
  const float code = rintf((a + 1.0f) * 0.5f * (float)(LEVELS - 1)); // RNE = jnp.round
  out_codes[(size_t)b * (1 + ACD) * TT + (size_t)(1 + d) * TT + t] = code;
  out_recon[(size_t)b * DD * TT + (size_t)(SEM + d) * TT + t] =
      code * (2.0f / (float)(LEVELS - 1)) - 1.0f;
}

// ---------------------------------------------------------------------------
// Kernel 3: fused VQ distance GEMM + argmin + gather.
//  - 4 waves/block, 16 flat-rows per wave (64 rows/block), 512 blocks
//  - A (x slab) kept VGPR-resident as bf16 hi/lo WMMA fragments (128 VGPR/lane)
//  - B (emb tile) in LDS, double-buffered, filled by async global->LDS DMA
//    (ASYNCcnt) and prefetched one tile ahead of compute: 1 barrier/tile
//  - dot via bf16x3 split (hi*hi + hi*lo + lo*hi), two f32 accumulators
//  - score = e2 - 2*dot (x2 is argmin-invariant)
// ---------------------------------------------------------------------------
#define MROWS 64
#define NTHR  128
#define BPAD  264        // bf16 elems per LDS row: 256 + 8 (528B = 33*16B: aligned, conflict-free)
#define ROWB  (BPAD * 2) // 528 bytes per LDS row
#define NTILES (NC / 16) // 512

__global__ __launch_bounds__(NTHR)
void vq_argmin_kernel(const float* __restrict__ x,
                      const float* __restrict__ emb_sum,
                      const float* __restrict__ usage,
                      const __bf16* __restrict__ ehi,
                      const float* __restrict__ e2g,
                      float* __restrict__ out_codes,
                      float* __restrict__ out_recon) {
  // [buf][plane(hi/lo)][row][col]
  __shared__ __align__(16) __bf16 Bbuf[2][2][16][BPAD];   // 33,792 B
  __shared__ int   codes_s[MROWS];
  __shared__ float uclamp_s[MROWS];

  const int tid  = threadIdx.x;
  const int lane = tid & 31;
  const int wave = tid >> 5;      // 0..3
  const int half = lane >> 4;     // lane half
  const int l16  = lane & 15;

  const int row0 = blockIdx.x * MROWS;   // flat row = b*T + t (block stays in one b)
  const int b  = row0 / TT;
  const int t0 = row0 % TT;

  const unsigned long long sbase = (unsigned long long)(const void*)ehi;
  const unsigned loPlaneOff = (unsigned)((size_t)NC * SEM * 2);   // elo = ehi + 4MB
  const unsigned bB[2] = { (unsigned)(size_t)(void*)&Bbuf[0][0][0][0],
                           (unsigned)(size_t)(void*)&Bbuf[1][0][0][0] };

  // ---- issue async DMA of codebook tile nt (16 rows x 512B, hi+lo) into buf ----
  // global plane rows are contiguous: byte offset = nt*8192 + chunk*16
  auto issue_tile = [&](int nt, int buf) {
#pragma unroll
    for (int i = 0; i < 4; ++i) {
      const int chunk = tid + i * NTHR;                     // 0..511
      const unsigned lds = bB[buf] + (unsigned)((chunk >> 5) * ROWB + (chunk & 31) * 16);
      const unsigned g   = (unsigned)(nt * 8192 + chunk * 16);
      async_ld16(lds,            g,              sbase);    // hi plane
      async_ld16(lds + 16 * ROWB, g + loPlaneOff, sbase);   // lo plane
    }
  };

  // ---- A fragments: VGPR-resident bf16 hi/lo for this wave's 16 rows ----
  // A layout (16-bit, 16x32): elems 0-7 -> K = ks*32 + half*8 + j
  //                           elems 8-15 -> K = ks*32 + 16 + half*8 + j
  bf16v a_hi[8], a_lo[8];
  {
    const int trow = t0 + wave * 16 + l16;
    const float* xb = x + (size_t)b * DD * TT + trow;
#pragma unroll
    for (int ks = 0; ks < 8; ++ks) {
      bf16v h, l;
#pragma unroll
      for (int j = 0; j < 8; ++j) {
        const float v0 = xb[(size_t)(ks * 32 + half * 8 + j) * TT];
        const float v1 = xb[(size_t)(ks * 32 + 16 + half * 8 + j) * TT];
        const __bf16 h0 = (__bf16)v0;
        const __bf16 h1 = (__bf16)v1;
        h[j] = h0;      l[j] = (__bf16)(v0 - (float)h0);
        h[8 + j] = h1;  l[8 + j] = (__bf16)(v1 - (float)h1);
      }
      a_hi[ks] = h;
      a_lo[ks] = l;
    }
  }

  float minv[8];
  int   mini[8];
#pragma unroll
  for (int r = 0; r < 8; ++r) { minv[r] = 3.4e38f; mini[r] = 0; }

  issue_tile(0, 0);                      // prefetch first tile

  for (int nt = 0; nt < NTILES; ++nt) {
    const int buf = nt & 1;
    wait_async0();                       // this wave's chunks of tile nt landed
    __syncthreads();                     // all waves' chunks visible; buf^1 free
    if (nt + 1 < NTILES) issue_tile(nt + 1, buf ^ 1);   // prefetch behind compute

    const float e2v = e2g[nt * 16 + l16];

    const __bf16* Bh = &Bbuf[buf][0][0][0];
    const __bf16* Bl = &Bbuf[buf][1][0][0];
    v8f acc0 = {}, acc1 = {};
#pragma unroll
    for (int ks = 0; ks < 8; ++ks) {
      // B layout (16-bit, 32x16): elems 0-7 -> K = ks*32 + half*8 + j, elems 8-15 -> K = ks*32 + 16 + half*8 + j of column n=l16
      const int off = l16 * BPAD + ks * 32 + half * 8;
      const bf16v bh = cat8(*(const bf8v*)&Bh[off], *(const bf8v*)&Bh[off + 16]);
      const bf16v bl = cat8(*(const bf8v*)&Bl[off], *(const bf8v*)&Bl[off + 16]);
      v8f& acc = (ks & 1) ? acc1 : acc0;   // two independent 12-deep WMMA chains
      acc = __builtin_amdgcn_wmma_f32_16x16x32_bf16(false, a_hi[ks], false, bh, (short)0, acc, false, false);
      acc = __builtin_amdgcn_wmma_f32_16x16x32_bf16(false, a_hi[ks], false, bl, (short)0, acc, false, false);
      acc = __builtin_amdgcn_wmma_f32_16x16x32_bf16(false, a_lo[ks], false, bh, (short)0, acc, false, false);
    }

    // C layout: lane -> N = l16; VGPR r -> M = r + 8*half
    const int ncode = nt * 16 + l16;
#pragma unroll
    for (int r = 0; r < 8; ++r) {
      const float score = e2v - 2.0f * (acc0[r] + acc1[r]);
      if (score < minv[r]) { minv[r] = score; mini[r] = ncode; }  // strict <: first index wins
    }
  }

  // argmin across the 16 lanes of each half (fixed M lives in one half)
#pragma unroll
  for (int r = 0; r < 8; ++r) {
#pragma unroll
    for (int off = 8; off >= 1; off >>= 1) {
      const float ov = __shfl_xor(minv[r], off, 32);
      const int   oi = __shfl_xor(mini[r], off, 32);
      if (ov < minv[r] || (ov == minv[r] && oi < mini[r])) { minv[r] = ov; mini[r] = oi; }
    }
  }
  if (l16 == 0) {
#pragma unroll
    for (int r = 0; r < 8; ++r) codes_s[wave * 16 + half * 8 + r] = mini[r];
  }
  __syncthreads();

  // sem codes (as f32) + clamped usage per row
  if (tid < MROWS) {
    const int c = codes_s[tid];
    out_codes[(size_t)b * (1 + ACD) * TT + (t0 + tid)] = (float)c;
    uclamp_s[tid] = fmaxf(usage[c], EPSV);
  }
  __syncthreads();

  // recon gather: recon[b][d][t0+m] = emb_sum[code[m]][d] / clamp(usage) ; t-coalesced writes
  {
    const int tl = tid & 63;
    const int dbase = tid >> 6;          // 2 d's per pass
    const int   c = codes_s[tl];
    const float u = uclamp_s[tl];
    for (int i = 0; i < SEM / 2; ++i) {
      const int d = dbase + i * 2;
      out_recon[(size_t)b * DD * TT + (size_t)d * TT + t0 + tl] =
          emb_sum[(size_t)c * SEM + d] / u;
    }
  }
}

// ---------------------------------------------------------------------------
extern "C" void kernel_launch(void* const* d_in, const int* in_sizes, int n_in,
                              void* d_out, int out_size, void* d_ws, size_t ws_size,
                              hipStream_t stream) {
  (void)in_sizes; (void)n_in; (void)out_size; (void)ws_size;
  const float* x       = (const float*)d_in[0];
  const float* emb_sum = (const float*)d_in[1];
  const float* usage   = (const float*)d_in[2];

  // outputs: codes (8*37*4096, small ints as f32) then recon (8*292*4096)
  float* out_codes = (float*)d_out;
  float* out_recon = out_codes + (size_t)BB * (1 + ACD) * TT;

  // workspace: bf16 hi plane (4MB) | bf16 lo plane (4MB, contiguous) | e2 (32KB)
  __bf16* ehi = (__bf16*)d_ws;
  __bf16* elo = ehi + (size_t)NC * SEM;
  float*  e2  = (float*)(elo + (size_t)NC * SEM);

  prep_emb_kernel<<<NC, SEM, 0, stream>>>(emb_sum, usage, ehi, elo, e2);

  const int acTotal = BB * ACD * TT;
  ac_kernel<<<(acTotal + 255) / 256, 256, 0, stream>>>(x, out_codes, out_recon);

  vq_argmin_kernel<<<(BB * TT) / MROWS, NTHR, 0, stream>>>(
      x, emb_sum, usage, ehi, e2, out_codes, out_recon);
}
